// SelfOrganizingMap_3066606649567
// MI455X (gfx1250) — matrix operand order repaired and checked by reading, not compiled, not run on hardware
//
#include <hip/hip_runtime.h>

#define M_CELLS 4096
#define DIM     256
#define BATCH   8192

typedef __attribute__((ext_vector_type(16))) __bf16         v16bf;
typedef __attribute__((ext_vector_type(8)))  float          v8f;
typedef __attribute__((ext_vector_type(16))) unsigned short v16u;
typedef __attribute__((ext_vector_type(4)))  int            v4i;

static __device__ __forceinline__ unsigned short f32_bf16(float f) {
  unsigned int u = __builtin_bit_cast(unsigned int, f);
  u += 0x7FFFu + ((u >> 16) & 1u);        // round-to-nearest-even
  return (unsigned short)(u >> 16);
}

// ---- async LDS copy (CDNA5 GLOBAL_LOAD_ASYNC_TO_LDS_B128), guarded fallback
#if defined(__has_builtin)
#if __has_builtin(__builtin_amdgcn_global_load_async_to_lds_b128) && \
    __has_builtin(__builtin_amdgcn_s_wait_asynccnt)
#define USE_ASYNC_LDS 1
#endif
#endif

static __device__ __forceinline__ void lds_copy16(const void* g, void* l) {
#if defined(USE_ASYNC_LDS)
  __builtin_amdgcn_global_load_async_to_lds_b128(
      (__attribute__((address_space(1))) v4i*)(void*)g,
      (__attribute__((address_space(3))) v4i*)l, 0, 0);
#else
  *(uint4*)l = *(const uint4*)g;
#endif
}
#if defined(USE_ASYNC_LDS)
#define LDS_COPY_WAIT(n) __builtin_amdgcn_s_wait_asynccnt(n)
#else
#define LDS_COPY_WAIT(n) ((void)0)
#endif

struct U128x2 { uint4 a, b; };   // 32B carrier for building v16bf from two b128 loads

// ---------------------------------------------------------------- zero ws
__global__ void k_zero(float* __restrict__ p, int n) {
  int i = blockIdx.x * 256 + threadIdx.x;
  if (i < n) p[i] = 0.0f;
}

// ---------------------------------------------------------------- f32 -> bf16 pre-pass
__global__ void k_cvt(const float* __restrict__ src, unsigned short* __restrict__ dst, int n4) {
  int i = blockIdx.x * 256 + threadIdx.x;
  if (i >= n4) return;
  const float4 v = ((const float4*)src)[i];
  ushort4 o;
  o.x = f32_bf16(v.x); o.y = f32_bf16(v.y); o.z = f32_bf16(v.z); o.w = f32_bf16(v.w);
  ((ushort4*)dst)[i] = o;
}

// ---------------------------------------------------------------- ||w_m||^2
__global__ void k_wnorm(const float* __restrict__ W, float* __restrict__ wn) {
  int m = blockIdx.x * 256 + threadIdx.x;
  if (m >= M_CELLS) return;
  const float4* p = (const float4*)(W + (size_t)m * DIM);
  float s = 0.0f;
  for (int i = 0; i < DIM / 4; ++i) {
    float4 v = p[i];
    s += v.x * v.x + v.y * v.y + v.z * v.z + v.w * v.w;
  }
  wn[m] = s;
}

// ---------------------------------------------------------------- BMU search
// block = 256 thr (8 waves), 64 batch rows/block; waves: g=row group, h=col half
// W tiles stream through a double-buffered LDS stage fed by async DMA.
#define LDSW_STRIDE 272   // 256 + 16 pad shorts; 32B-aligned rows, fewer bank conflicts
#define NCHW (M_CELLS / 32)
__global__ void __launch_bounds__(256) k_bmu(const unsigned short* __restrict__ Xb,
                                             const unsigned short* __restrict__ Wb,
                                             const float* __restrict__ wnorm,
                                             int*   __restrict__ bmuWS,
                                             float* __restrict__ outLocs) {
  __shared__ __align__(32) unsigned short ldsW[2][32 * LDSW_STRIDE];
  __shared__ float sVal[4][2][16];
  __shared__ int   sIdx[4][2][16];

  const int tid  = threadIdx.x;
  const int wave = tid >> 5;
  const int lane = tid & 31;
  const int g  = wave & 3;
  const int h  = wave >> 2;
  const int hi = lane >> 4;

  // A fragments: 16 X rows (bf16) in registers, reused for all column chunks.
  const int rowA = blockIdx.x * 64 + g * 16 + (lane & 15);
  const unsigned short* xp = Xb + (size_t)rowA * DIM;
  v16bf a[8];
  for (int kc = 0; kc < 8; ++kc) {
    const int k0 = kc * 32 + (hi ? 16 : 0); // lanes 0-15: K 0-15; 16-31: 16-31
    U128x2 u;
    u.a = *(const uint4*)(xp + k0);
    u.b = *(const uint4*)(xp + k0 + 8);
    a[kc] = __builtin_bit_cast(v16bf, u);
  }

  float runVal[8];
  int   runIdx[8];
  for (int r = 0; r < 8; ++r) { runVal[r] = 3.4e38f; runIdx[r] = 0; }

  const int mloc = h * 16 + (lane & 15);
  const int koff = hi ? 16 : 0;

  // per-thread staging decomposition, hoisted out of the chunk loop
  int mmv[4], prt[4];
  for (int it = 0; it < 4; ++it) {
    const int lin = it * 256 + tid;         // 0..1023 = 32 rows x 32 granules
    mmv[it] = lin >> 5;
    prt[it] = (lin & 31) * 8;
  }
  auto issueW = [&](int c, int buf) {
    for (int it = 0; it < 4; ++it)
      lds_copy16(Wb + (size_t)(c * 32 + mmv[it]) * DIM + prt[it],
                 &ldsW[buf][mmv[it] * LDSW_STRIDE + prt[it]]);
  };

  issueW(0, 0);                             // prime the pipeline
  for (int ch = 0; ch < NCHW; ++ch) {
    const int buf = ch & 1;
    if (ch + 1 < NCHW) {
      issueW(ch + 1, buf ^ 1);              // DMA next tile under this tile's WMMA
      LDS_COPY_WAIT(4);                     // in-order: the 4 older copies (this tile) done
    } else {
      LDS_COPY_WAIT(0);
    }
    __syncthreads();

    v8f c = {};
    for (int kc = 0; kc < 8; ++kc) {
      const v16u* bp = (const v16u*)&ldsW[buf][mloc * LDSW_STRIDE + kc * 32 + koff];
      v16bf b = __builtin_bit_cast(v16bf, *bp);
      c = __builtin_amdgcn_wmma_f32_16x16x32_bf16(false, a[kc], false, b,
                                                  (short)0, c, false, false);
    }
    const int   m  = ch * 32 + mloc;
    const float wn = wnorm[m];
    for (int r = 0; r < 8; ++r) {
      float s = wn - 2.0f * c[r];           // argmin-equivalent squared distance
      if (s < runVal[r]) { runVal[r] = s; runIdx[r] = m; }
    }
    __syncthreads();                        // all reads of buf done before it is re-filled
  }

  // min-reduce across the 16-lane C-tile groups (masks 1..8 stay in-group)
  for (int msk = 1; msk <= 8; msk <<= 1) {
    for (int r = 0; r < 8; ++r) {
      float ov = __shfl_xor(runVal[r], msk, 32);
      int   oi = __shfl_xor(runIdx[r], msk, 32);
      if (ov < runVal[r] || (ov == runVal[r] && oi < runIdx[r])) {
        runVal[r] = ov; runIdx[r] = oi;
      }
    }
  }
  if (lane == 0)
    for (int r = 0; r < 8; ++r) { sVal[g][h][r] = runVal[r]; sIdx[g][h][r] = runIdx[r]; }
  if (lane == 16)
    for (int r = 0; r < 8; ++r) { sVal[g][h][8 + r] = runVal[r]; sIdx[g][h][8 + r] = runIdx[r]; }
  __syncthreads();

  if (tid < 64) {     // combine the two column-half waves per row
    const int gg = tid >> 4, rr = tid & 15;
    float v0 = sVal[gg][0][rr]; int i0 = sIdx[gg][0][rr];
    float v1 = sVal[gg][1][rr]; int i1 = sIdx[gg][1][rr];
    int best = (v1 < v0 || (v1 == v0 && i1 < i0)) ? i1 : i0;
    int b = blockIdx.x * 64 + tid;
    bmuWS[b] = best;
    outLocs[2 * b]     = (float)(best >> 6);
    outLocs[2 * b + 1] = (float)(best & 63);
  }
}

// ---------------------------------------------------------------- update GEMM
// numerator[m,d] += sum_b lr[b,m] * x[b,d];  lr generated on the fly.
#define LDSX_STRIDE 48    // 32 + 16 pad shorts, 32B-aligned rows
__global__ void __launch_bounds__(256) k_update(const unsigned short* __restrict__ Xb,
                                                const int*   __restrict__ bmuWS,
                                                const int*   __restrict__ epoch_p,
                                                float* __restrict__ numerWS,
                                                float* __restrict__ denomWS) {
  __shared__ __align__(32) unsigned short ldsX[DIM * LDSX_STRIDE];

  const int tid  = threadIdx.x;
  const int wave = tid >> 5;
  const int lane = tid & 31;
  const int g  = wave & 3;
  const int h  = wave >> 2;
  const int hi = lane >> 4;
  const int m0 = blockIdx.x * 64;

  const float ep     = (float)epoch_p[0];
  const float radius = 32.0f - ep * (31.0f / 99.0f);
  const float alpha  = 0.1f * (1.0f - ep * 0.01f);
  const float sigma  = radius * 0.5f;
  const float sc     = -1.0f / (2.0f * sigma * sigma);

  const int mrow = m0 + g * 16 + (lane & 15);
  const int mr = mrow >> 6, mc = mrow & 63;

  v8f acc[8];
  for (int t = 0; t < 8; ++t) acc[t] = (v8f){};
  float denomAcc = 0.0f;

  const int bstart = blockIdx.y * (BATCH / 8);
  for (int ch = 0; ch < (BATCH / 8) / 32; ++ch) {
    const int kbase = bstart + ch * 32;
    // transpose-stage X chunk to [d][b] bf16 (coalesced bf16 global reads)
    for (int bb = 0; bb < 32; ++bb)
      ldsX[tid * LDSX_STRIDE + bb] = Xb[(size_t)(kbase + bb) * DIM + tid];
    __syncthreads();

    // synthesize A fragment: lr tile [16 m x 32 b]
    v16u ua;
    for (int j = 0; j < 16; ++j) {
      const int boff = j + (hi ? 16 : 0);
      const int bi = bmuWS[kbase + boff];
      const int dr = mr - (bi >> 6);
      const int dc = mc - (bi & 63);
      const float lr = alpha * __expf(sc * (float)(dr * dr + dc * dc));
      if (h == 0) denomAcc += lr;           // count each (m,b) exactly once
      ua[j] = f32_bf16(lr);
    }
    v16bf afrag = __builtin_bit_cast(v16bf, ua);

    for (int nt = 0; nt < 8; ++nt) {
      const int dcol = h * 128 + nt * 16 + (lane & 15);
      const v16u* bp = (const v16u*)&ldsX[dcol * LDSX_STRIDE + (hi ? 16 : 0)];
      v16bf bfrag = __builtin_bit_cast(v16bf, *bp);
      acc[nt] = __builtin_amdgcn_wmma_f32_16x16x32_bf16(false, afrag, false, bfrag,
                                                        (short)0, acc[nt], false, false);
    }
    __syncthreads();
  }

  if (h == 0) {
    denomAcc += __shfl_xor(denomAcc, 16, 32);
    if (lane < 16) atomicAdd(&denomWS[m0 + g * 16 + lane], denomAcc);
  }
  for (int nt = 0; nt < 8; ++nt) {
    const int dcol = h * 128 + nt * 16 + (lane & 15);
    for (int r = 0; r < 8; ++r) {
      const int mm = m0 + g * 16 + r + (hi ? 8 : 0);
      atomicAdd(&numerWS[(size_t)mm * DIM + dcol], acc[nt][r]);
    }
  }
}

// ---------------------------------------------------------------- finalize
__global__ void k_final(const float* __restrict__ numer,
                        const float* __restrict__ denom,
                        float* __restrict__ out) {
  int i = blockIdx.x * 256 + threadIdx.x;
  if (i < M_CELLS * DIM) out[i] = numer[i] / (denom[i >> 8] + 1e-12f);
}

// ---------------------------------------------------------------- launcher
extern "C" void kernel_launch(void* const* d_in, const int* in_sizes, int n_in,
                              void* d_out, int out_size, void* d_ws, size_t ws_size,
                              hipStream_t stream) {
  (void)in_sizes; (void)n_in; (void)out_size; (void)ws_size;
  const float* X     = (const float*)d_in[0];
  const float* W     = (const float*)d_in[1];
  const int*   epoch = (const int*)d_in[2];
  float* out = (float*)d_out;

  char*  ws      = (char*)d_ws;
  float* wnorm   = (float*)ws;                          // 4096 f32
  int*   bmuWS   = (int*)(ws + 16384);                  // 8192 i32
  float* denomWS = (float*)(ws + 49152);                // 4096 f32
  float* numerWS = (float*)(ws + 65536);                // 4096*256 f32
  unsigned short* Xb = (unsigned short*)(ws + 4259840); // 8192*256 bf16 (4 MB)
  unsigned short* Wb = (unsigned short*)(ws + 8454144); // 4096*256 bf16 (2 MB)

  const int nz = M_CELLS + M_CELLS * DIM;               // zero denom + numer in one pass
  k_zero<<<(nz + 255) / 256, 256, 0, stream>>>(denomWS, nz);
  k_cvt<<<(BATCH * DIM / 4 + 255) / 256, 256, 0, stream>>>(X, Xb, BATCH * DIM / 4);
  k_cvt<<<(M_CELLS * DIM / 4 + 255) / 256, 256, 0, stream>>>(W, Wb, M_CELLS * DIM / 4);
  k_wnorm<<<(M_CELLS + 255) / 256, 256, 0, stream>>>(W, wnorm);
  k_bmu<<<BATCH / 64, 256, 0, stream>>>(Xb, Wb, wnorm, bmuWS, out);
  k_update<<<dim3(M_CELLS / 64, 8), 256, 0, stream>>>(Xb, bmuWS, epoch, numerWS, denomWS);
  k_final<<<(M_CELLS * DIM + 255) / 256, 256, 0, stream>>>(numerWS, denomWS, out + 2 * BATCH);
}
